// GNN_11003706213252
// MI455X (gfx1250) — matrix operand order repaired and checked by reading, not compiled, not run on hardware
//
#include <hip/hip_runtime.h>
#include <hip/hip_bf16.h>
#include <math.h>

typedef __attribute__((ext_vector_type(16))) _Float16 v16h;
typedef __attribute__((ext_vector_type(8)))  _Float16 v8h;
typedef __attribute__((ext_vector_type(8)))  float    v8f;
typedef __attribute__((ext_vector_type(4)))  float    v4f;

#define IN_DIM   128
#define HID      64
#define HEADS    4
#define OUT_DIM  64
#define QKV_C    768            // 3 * 256
#define WALL_C   832            // 768 + 64 (skip)
#define NTILE1   52             // WALL_C / 16
#define SKIP_T0  48             // first skip column tile

// ---------------------------------------------------------------------------
// helpers
// ---------------------------------------------------------------------------
__device__ __forceinline__ void atomicMaxF(float* addr, float val) {
    // sign-split trick; init value must be -inf.
    if (val >= 0.0f) atomicMax((int*)addr, __float_as_int(val));
    else             atomicMin((unsigned int*)addr, __float_as_uint(val));
}

__device__ __forceinline__ v16h cat16(v8h lo, v8h hi) {
    return __builtin_shufflevector(lo, hi, 0, 1, 2, 3, 4, 5, 6, 7,
                                           8, 9, 10, 11, 12, 13, 14, 15);
}

// ---------------------------------------------------------------------------
// prep kernels
// ---------------------------------------------------------------------------
__global__ void gat_prep_x(const float* __restrict__ x, _Float16* __restrict__ xh, int n) {
    int i = blockIdx.x * blockDim.x + threadIdx.x;
    if (i < n) xh[i] = (_Float16)x[i];
}

// Builds Wt: TRANSPOSED fused weights, Wt[col * 128 + k], col in [0,832)
__global__ void gat_prep_w(const float* __restrict__ Wq, const float* __restrict__ Wk,
                           const float* __restrict__ Wv, const float* __restrict__ Wskip,
                           const float* __restrict__ bq, const float* __restrict__ bk,
                           const float* __restrict__ bv, const float* __restrict__ bskip,
                           _Float16* __restrict__ Wt, float* __restrict__ ball) {
    int idx = blockIdx.x * blockDim.x + threadIdx.x;
    int total = IN_DIM * WALL_C;
    if (idx < total) {
        int i = idx / WALL_C;          // k (input dim)
        int j = idx - i * WALL_C;      // fused output col
        float w;
        if      (j < 256) w = Wq[i * 256 + j];
        else if (j < 512) w = Wk[i * 256 + (j - 256)];
        else if (j < 768) w = Wv[i * 256 + (j - 512)];
        else              w = Wskip[i * 64 + (j - 768)];
        Wt[(size_t)j * IN_DIM + i] = (_Float16)w;   // column-major store
    }
    if (idx < WALL_C) {
        float b;
        if      (idx < 256) b = bq[idx];
        else if (idx < 512) b = bk[idx - 256];
        else if (idx < 768) b = bv[idx - 512];
        else                b = bskip[idx - 768];
        ball[idx] = b;
    }
}

__global__ void gat_prep_init(float* __restrict__ m, float* __restrict__ denom, int n4) {
    int i = blockIdx.x * blockDim.x + threadIdx.x;
    if (i < n4) { m[i] = -__builtin_inff(); denom[i] = 0.0f; }
}

// ---------------------------------------------------------------------------
// GEMM 1: [N x 128] @ [128 x 832] via v_wmma_f32_16x16x32_f16.
// 8 waves per block, one 16x16 tile per wave, K-loop of 4 WMMAs.
// A fragment: lanes 0-15 = rows, K runs [hi*8,+8) and [16+hi*8,+8)  (ISA 7.12.2)
// B fragment: lane&15 = col, K runs [hi*8,+8) and [16+hi*8,+8), same as A   (ISA 7.12.4)
// Epilogue branches are wave-uniform (per-tile), no exec masking.
// ---------------------------------------------------------------------------
__global__ __launch_bounds__(256)
void gat_gemm_qkvs(const _Float16* __restrict__ xh, const _Float16* __restrict__ Wt,
                   const float* __restrict__ ball, _Float16* __restrict__ qkv,
                   float* __restrict__ nodeacc, int n, int ntiles) {
    const int wave = threadIdx.x >> 5;
    const int lane = threadIdx.x & 31;
    const int tile = blockIdx.x * 8 + wave;
    if (tile >= ntiles) return;                 // wave-uniform exit
    const int tm = tile / NTILE1;
    const int tn = tile - tm * NTILE1;
    const int lsel = lane & 15;
    const int hi   = lane >> 4;

    int row = tm * 16 + lsel;
    if (row >= n) row = n - 1;                  // branch-free clamp for loads
    const _Float16* arow = xh + (size_t)row * IN_DIM;
    const _Float16* bcol = Wt + (size_t)(tn * 16 + lsel) * IN_DIM;

    v8f c = {};
    #pragma unroll
    for (int kb = 0; kb < IN_DIM; kb += 32) {
        if (kb + 32 < IN_DIM) {
            __builtin_prefetch(arow + kb + 32, 0, 3);
            __builtin_prefetch(bcol + kb + 32, 0, 3);
        }
        v8h alo = *(const v8h*)(arow + kb + hi * 8);
        v8h ahi = *(const v8h*)(arow + kb + 16 + hi * 8);
        v8h blo = *(const v8h*)(bcol + kb + hi * 8);
        v8h bhi = *(const v8h*)(bcol + kb + 16 + hi * 8);
        c = __builtin_amdgcn_wmma_f32_16x16x32_f16(false, cat16(alo, ahi),
                                                   false, cat16(blo, bhi),
                                                   (short)0, c, false, false);
    }

    // D layout: VGPR r, lanes 0-15 -> M=r, lanes 16-31 -> M=r+8; N = lane&15.
    const int gcol = tn * 16 + lsel;
    const float bias = ball[gcol];
    const int mbase = tm * 16 + hi * 8;
    const bool fullTile = (tm * 16 + 16 <= n);   // uniform; always true for N%16==0

    if (tn < SKIP_T0) {                          // uniform: q/k/v columns -> f16
        _Float16* qp = qkv + (size_t)mbase * QKV_C + gcol;
        if (fullTile) {
            #pragma unroll
            for (int r = 0; r < 8; ++r)
                qp[(size_t)r * QKV_C] = (_Float16)(c[r] + bias);
        } else {
            #pragma unroll
            for (int r = 0; r < 8; ++r)
                if (mbase + r < n) qp[(size_t)r * QKV_C] = (_Float16)(c[r] + bias);
        }
    } else {                                     // uniform: skip columns -> f32 accum
        float* ap = nodeacc + (size_t)mbase * 64 + (gcol - QKV_C);
        if (fullTile) {
            #pragma unroll
            for (int r = 0; r < 8; ++r)
                ap[(size_t)r * 64] = c[r] + bias;
        } else {
            #pragma unroll
            for (int r = 0; r < 8; ++r)
                if (mbase + r < n) ap[(size_t)r * 64] = c[r] + bias;
        }
    }
}

// ---------------------------------------------------------------------------
// Edge kernels
// ---------------------------------------------------------------------------
__global__ void gat_edge_alpha(const int* __restrict__ ei, const _Float16* __restrict__ qkv,
                               float* __restrict__ alpha, float* __restrict__ m, int E) {
    int idx = blockIdx.x * blockDim.x + threadIdx.x;
    if (idx >= E * HEADS) return;
    int e = idx >> 2, h = idx & 3;
    int src = ei[e], dst = ei[E + e];
    const v8h* q8 = (const v8h*)(qkv + (size_t)dst * QKV_C + h * HID);        // Q
    const v8h* k8 = (const v8h*)(qkv + (size_t)src * QKV_C + 256 + h * HID);  // K
    float s = 0.0f;
    #pragma unroll
    for (int d = 0; d < 8; ++d) {
        v8h qa = q8[d], ka = k8[d];
        #pragma unroll
        for (int j = 0; j < 8; ++j) s += (float)qa[j] * (float)ka[j];
    }
    s *= 0.125f;   // 1/sqrt(64)
    alpha[idx] = s;
    atomicMaxF(&m[dst * HEADS + h], s);
}

__global__ void gat_edge_exp(const int* __restrict__ ei, float* __restrict__ alpha,
                             const float* __restrict__ m, float* __restrict__ denom, int E) {
    int idx = blockIdx.x * blockDim.x + threadIdx.x;
    if (idx >= E * HEADS) return;
    int e = idx >> 2, h = idx & 3;
    int dst = ei[E + e];
    float ex = __expf(alpha[idx] - m[dst * HEADS + h]);
    alpha[idx] = ex;                       // reuse buffer: now holds exp()
    atomicAdd(&denom[dst * HEADS + h], ex);
}

__global__ void gat_edge_msg(const int* __restrict__ ei, const _Float16* __restrict__ qkv,
                             const float* __restrict__ alpha, const float* __restrict__ denom,
                             float* __restrict__ nodeacc, int E) {
    int idx = blockIdx.x * blockDim.x + threadIdx.x;
    if (idx >= E * HID) return;
    int e = idx >> 6, d = idx & 63;
    int src = ei[e], dst = ei[E + e];
    const _Float16* vr = qkv + (size_t)src * QKV_C + 512 + d;   // V block, element d
    float acc = 0.0f;
    #pragma unroll
    for (int h = 0; h < HEADS; ++h) {
        float a = alpha[e * HEADS + h] / (denom[dst * HEADS + h] + 1e-16f);
        acc += a * (float)vr[h * HID];
    }
    // 0.25 = head mean (concat=False)
    atomicAdd(&nodeacc[(size_t)dst * 64 + d], acc * 0.25f);
}

// ---------------------------------------------------------------------------
// Spectral norm (64x64, 20 power iterations) + scale, stored TRANSPOSED:
// Wsct[col * 64 + k] = Wmlp[k][col] / sigma
// ---------------------------------------------------------------------------
__device__ __forceinline__ float blk_reduce_sum64(float x, float* red) {
    int t = threadIdx.x;
    red[t] = x; __syncthreads();
    #pragma unroll
    for (int s = 32; s > 0; s >>= 1) {
        if (t < s) red[t] += red[t + s];
        __syncthreads();
    }
    float r = red[0]; __syncthreads();
    return r;
}

__global__ __launch_bounds__(64)
void gat_sigma_scale(const float* __restrict__ W, _Float16* __restrict__ Wsct) {
    __shared__ float u[64], vv[64], red[64];
    int t = threadIdx.x;
    u[t] = 0.125f;                 // ones / ||ones||, ||ones|| = 8
    __syncthreads();
    for (int it = 0; it < 20; ++it) {
        float s = 0.0f;
        for (int i = 0; i < 64; ++i) s += W[i * 64 + t] * u[i];
        float nv = sqrtf(blk_reduce_sum64(s * s, red)) + 1e-12f;
        vv[t] = s / nv; __syncthreads();
        float s2 = 0.0f;
        for (int j = 0; j < 64; ++j) s2 += W[t * 64 + j] * vv[j];
        float nu = sqrtf(blk_reduce_sum64(s2 * s2, red)) + 1e-12f;
        u[t] = s2 / nu; __syncthreads();
    }
    float s2 = 0.0f;
    for (int j = 0; j < 64; ++j) s2 += W[t * 64 + j] * vv[j];
    float sigma = blk_reduce_sum64(u[t] * s2, red);
    float inv = 1.0f / sigma;
    for (int idx = t; idx < 64 * 64; idx += 64) {
        int k = idx >> 6, col = idx & 63;
        Wsct[col * 64 + k] = (_Float16)(W[idx] * inv);
    }
}

// ---------------------------------------------------------------------------
// GEMM 2: out = tanh(nodeacc) @ (Wmlp / sigma), [N x 64] @ [64 x 64]
// ---------------------------------------------------------------------------
__global__ __launch_bounds__(256)
void gat_final_gemm(const float* __restrict__ nodeacc, const _Float16* __restrict__ Wsct,
                    float* __restrict__ out, int n, int ntiles) {
    const int wave = threadIdx.x >> 5;
    const int lane = threadIdx.x & 31;
    const int tile = blockIdx.x * 8 + wave;
    if (tile >= ntiles) return;
    const int tm = tile >> 2;          // 4 column tiles
    const int tn = tile & 3;
    const int lsel = lane & 15;
    const int hi   = lane >> 4;

    int row = tm * 16 + lsel;
    if (row >= n) row = n - 1;
    const float*    arow = nodeacc + (size_t)row * 64;
    const _Float16* bcol = Wsct + (size_t)(tn * 16 + lsel) * 64;

    v8f c = {};
    #pragma unroll
    for (int kb = 0; kb < 64; kb += 32) {
        v4f a0 = *(const v4f*)(arow + kb + hi * 8);
        v4f a1 = *(const v4f*)(arow + kb + hi * 8 + 4);
        v4f a2 = *(const v4f*)(arow + kb + 16 + hi * 8);
        v4f a3 = *(const v4f*)(arow + kb + 16 + hi * 8 + 4);
        v16h a;
        #pragma unroll
        for (int j = 0; j < 4; ++j) {
            a[j]      = (_Float16)tanhf(a0[j]);
            a[4 + j]  = (_Float16)tanhf(a1[j]);
            a[8 + j]  = (_Float16)tanhf(a2[j]);
            a[12 + j] = (_Float16)tanhf(a3[j]);
        }
        v8h blo = *(const v8h*)(bcol + kb + hi * 8);
        v8h bhi = *(const v8h*)(bcol + kb + 16 + hi * 8);
        c = __builtin_amdgcn_wmma_f32_16x16x32_f16(false, a, false, cat16(blo, bhi),
                                                   (short)0, c, false, false);
    }

    const int gcol  = tn * 16 + lsel;
    const int mbase = tm * 16 + hi * 8;
    float* op = out + (size_t)mbase * 64 + gcol;
    if (tm * 16 + 16 <= n) {                     // uniform fast path
        #pragma unroll
        for (int r = 0; r < 8; ++r) op[(size_t)r * 64] = c[r];
    } else {
        #pragma unroll
        for (int r = 0; r < 8; ++r)
            if (mbase + r < n) op[(size_t)r * 64] = c[r];
    }
}

// ---------------------------------------------------------------------------
// launcher
// ---------------------------------------------------------------------------
extern "C" void kernel_launch(void* const* d_in, const int* in_sizes, int n_in,
                              void* d_out, int out_size, void* d_ws, size_t ws_size,
                              hipStream_t stream) {
    const float* x     = (const float*)d_in[0];
    const int*   ei    = (const int*)  d_in[1];
    const float* Wq    = (const float*)d_in[2];
    const float* bq    = (const float*)d_in[3];
    const float* Wk    = (const float*)d_in[4];
    const float* bk    = (const float*)d_in[5];
    const float* Wv    = (const float*)d_in[6];
    const float* bv    = (const float*)d_in[7];
    const float* Wskip = (const float*)d_in[8];
    const float* bskip = (const float*)d_in[9];
    const float* Wmlp  = (const float*)d_in[10];

    const int n = in_sizes[0] / IN_DIM;     // 50000
    const int E = in_sizes[1] / 2;          // 300000

    // workspace carve-up (256B aligned slices)
    char* wsp = (char*)d_ws;
    size_t off = 0;
    auto carve = [&](size_t bytes) -> void* {
        void* p = wsp + off;
        off = (off + bytes + 255) & ~(size_t)255;
        return p;
    };
    _Float16* xh      = (_Float16*)carve((size_t)n * IN_DIM * sizeof(_Float16));
    _Float16* Wt      = (_Float16*)carve((size_t)IN_DIM * WALL_C * sizeof(_Float16));
    float*    ball    = (float*)   carve(WALL_C * sizeof(float));
    _Float16* qkv     = (_Float16*)carve((size_t)n * QKV_C * sizeof(_Float16));
    float*    nodeacc = (float*)   carve((size_t)n * 64 * sizeof(float));
    float*    alpha   = (float*)   carve((size_t)E * HEADS * sizeof(float));
    float*    mbuf    = (float*)   carve((size_t)n * HEADS * sizeof(float));
    float*    denom   = (float*)   carve((size_t)n * HEADS * sizeof(float));
    _Float16* Wsct    = (_Float16*)carve(64 * 64 * sizeof(_Float16));
    (void)ws_size;

    // prep
    {
        int tot = n * IN_DIM;
        gat_prep_x<<<(tot + 255) / 256, 256, 0, stream>>>(x, xh, tot);
        int totw = IN_DIM * WALL_C;
        gat_prep_w<<<(totw + 255) / 256, 256, 0, stream>>>(Wq, Wk, Wv, Wskip,
                                                           bq, bk, bv, bskip, Wt, ball);
        int n4 = n * HEADS;
        gat_prep_init<<<(n4 + 255) / 256, 256, 0, stream>>>(mbuf, denom, n4);
    }

    // GEMM 1: qkv + skip-into-accumulator
    {
        int mtiles = (n + 15) / 16;
        int ntiles = mtiles * NTILE1;            // 3125 * 52 tiles, 8 waves/block
        gat_gemm_qkvs<<<(ntiles + 7) / 8, 256, 0, stream>>>(xh, Wt, ball, qkv,
                                                            nodeacc, n, ntiles);
    }

    // attention: logits+max, exp+denom, messages
    {
        int eh = E * HEADS;
        gat_edge_alpha<<<(eh + 255) / 256, 256, 0, stream>>>(ei, qkv, alpha, mbuf, E);
        gat_edge_exp  <<<(eh + 255) / 256, 256, 0, stream>>>(ei, alpha, mbuf, denom, E);
        int ed = E * HID;
        gat_edge_msg  <<<(ed + 255) / 256, 256, 0, stream>>>(ei, qkv, alpha, denom,
                                                             nodeacc, E);
    }

    // spectral norm + scale (transposed output)
    gat_sigma_scale<<<1, 64, 0, stream>>>(Wmlp, Wsct);

    // GEMM 2: tanh + MLP
    {
        int ntiles = ((n + 15) / 16) * 4;        // 12500 tiles
        gat_final_gemm<<<(ntiles + 7) / 8, 256, 0, stream>>>(nodeacc, Wsct,
                                                             (float*)d_out, n, ntiles);
    }
}
